// BinarizedInputNetwork_14345190768700
// MI455X (gfx1250) — 8-bit matrix operand order repaired and checked by reading, compiled, not run on hardware
//
#include <hip/hip_runtime.h>
#include <stdint.h>

typedef __attribute__((ext_vector_type(8))) int v8i;
typedef __attribute__((ext_vector_type(4))) int v4i;

#ifndef __has_builtin
#define __has_builtin(x) 0
#endif
#if __has_builtin(__builtin_amdgcn_global_load_async_to_lds_b128) && \
    __has_builtin(__builtin_amdgcn_s_wait_asynccnt)
#define USE_ASYNC_COPY 1
typedef __attribute__((address_space(1))) v4i gas_v4i;
typedef __attribute__((address_space(3))) v4i las_v4i;
#else
#define USE_ASYNC_COPY 0
#endif

#define WS_BN_SCALE   0u
#define WS_BN_SHIFT   3840u
#define WS_POOLED     7680u
#define WS_W2         131072u
#define WS_W3         204800u
#define WS_W4         352256u
#define WS_W5         573440u
#define WS_A2         1048576u
#define WS_A3         34603008u
#define WS_A4         101711872u
#define WS_A5         118489088u

__device__ __forceinline__ int permch(int c)
{
    int dw = (c >> 2) & 15;
    int np = ((dw >> 1) & 1) * 8 + (dw >> 2) * 2 + (dw & 1);
    return (c & ~63) | (np << 2) | (c & 3);
}

__global__ void bn_prep_kernel(
    const float* g1, const float* b1, const float* m1, const float* v1,
    const float* g2, const float* b2, const float* m2, const float* v2,
    const float* g3, const float* b3, const float* m3, const float* v3,
    const float* g4, const float* b4, const float* m4, const float* v4,
    const float* g5, const float* b5, const float* m5, const float* v5,
    float* scale, float* shift)
{
    int tid = blockIdx.x * blockDim.x + threadIdx.x;
    if (tid >= 5 * 192) return;
    int layer = tid / 192, ch = tid % 192;
    const int   C[5] = {64, 128, 128, 192, 192};
    if (ch >= C[layer]) return;
    const float* G[5] = {g1, g2, g3, g4, g5};
    const float* B[5] = {b1, b2, b3, b4, b5};
    const float* M[5] = {m1, m2, m3, m4, m5};
    const float* V[5] = {v1, v2, v3, v4, v5};
    float inv = G[layer][ch] * rsqrtf(V[layer][ch] + 1e-5f);
    scale[layer * 192 + ch] = inv;
    shift[layer * 192 + ch] = B[layer][ch] - M[layer][ch] * inv;
}

__global__ void pack_weights_kernel(const float* __restrict__ w, int8_t* __restrict__ dst,
                                    int Cin, int Cout, int taps)
{
    int KB = Cin >> 6;
    int NT = Cout >> 4;
    int nfrag = taps * taps * KB * NT;
    int tid  = blockIdx.x * blockDim.x + threadIdx.x;
    int lane = tid & 31;
    int frag = tid >> 5;
    if (frag >= nfrag) return;
    int ct  = frag % NT;
    int rem = frag / NT;
    int kb  = rem % KB;
    int t   = rem / KB;
    int kh = t / taps, kw = t % taps;
    int khalf = lane >> 4, nl = lane & 15;
    int cout = ct * 16 + nl;
    uint32_t* out = (uint32_t*)(dst + (size_t)frag * 1024 + (size_t)lane * 32);
    for (int d = 0; d < 8; ++d) {
        uint32_t pk = 0;
        for (int bb = 0; bb < 4; ++bb) {
            int kl  = ((d >> 1) * 16) + khalf * 8 + (d & 1) * 4 + bb;
            int cin = kb * 64 + kl;
            float wv = w[(((size_t)cout * Cin + cin) * taps + kh) * taps + kw];
            int sv = (wv > 0.f) ? 1 : ((wv < 0.f) ? -1 : 0);
            pk |= ((uint32_t)(uint8_t)(int8_t)sv) << (8 * bb);
        }
        out[d] = pk;
    }
}

__global__ void __launch_bounds__(256)
conv1_kernel(const float* __restrict__ x, const float* __restrict__ w,
             const float* __restrict__ bias, const float* __restrict__ scale,
             const float* __restrict__ shift, int8_t* __restrict__ a2)
{
    __shared__ float sw[576], sb[64], ssc[64], ssh[64];
    int tid = threadIdx.x;
    for (int i = tid; i < 576; i += 256) sw[i] = w[i];
    if (tid < 64) { sb[tid] = bias[tid]; ssc[tid] = scale[tid]; ssh[tid] = shift[tid]; }
    __syncthreads();
    int gid = blockIdx.x * 256 + tid;
    int n = gid >> 12, h = (gid >> 6) & 63, wo = gid & 63;
    const float* xp = x + (size_t)n * 128 * 128;
    float in[9];
    #pragma unroll
    for (int kh = 0; kh < 3; ++kh)
        #pragma unroll
        for (int kw = 0; kw < 3; ++kw) {
            int hi = 2 * h + kh - 1, wi = 2 * wo + kw - 1;
            in[kh * 3 + kw] = (hi >= 0 && hi < 128 && wi >= 0 && wi < 128)
                                  ? xp[hi * 128 + wi] : 0.f;
        }
    int8_t* op = a2 + (size_t)gid * 64;
    for (int c = 0; c < 64; ++c) {
        float acc = sb[c];
        #pragma unroll
        for (int t = 0; t < 9; ++t) acc = fmaf(sw[c * 9 + t], in[t], acc);
        float y = acc * ssc[c] + ssh[c];
        op[permch(c)] = (y > 0.f) ? (int8_t)1 : (int8_t)0;
    }
}

template <int Hin, int Win, int Cin, int Hout, int Wout, int Cout,
          int STRIDE, int PAD, int TAPS, bool LAST>
__global__ void __launch_bounds__(256)
binconv_wmma(const int8_t* __restrict__ act_in, int8_t* __restrict__ act_out,
             const int8_t* __restrict__ wpack,
             const float* __restrict__ scale, const float* __restrict__ shift,
             float* __restrict__ pooled)
{
    constexpr int NT    = Cout / 16;
    constexpr int WT    = Wout / 16;
    constexpr int KB    = Cin / 64;
    constexpr int CHUNK = KB * NT * 1024;

    __shared__ int8_t sB[CHUNK];

    const int lane = threadIdx.x & 31;
    int tile = blockIdx.x * 8 + (threadIdx.x >> 5);
    const int wt = tile % WT;
    const int h  = (tile / WT) % Hout;
    const int n  = tile / (WT * Hout);

    const int m     = lane & 15;
    const int khalf = lane >> 4;
    const int w0    = wt * 16;

    v8i acc[NT];
    #pragma unroll
    for (int i = 0; i < NT; ++i) acc[i] = (v8i){0, 0, 0, 0, 0, 0, 0, 0};

    for (int kh = 0; kh < TAPS; ++kh) {
        int h_in = h * STRIDE + kh - PAD;
        bool h_ok = (h_in >= 0) && (h_in < Hin);
        const int8_t* rowbase =
            act_in + ((size_t)(n * Hin + (h_ok ? h_in : 0)) * Win) * Cin;
        for (int kw = 0; kw < TAPS; ++kw) {
            __syncthreads();
            {
                const int8_t* gsrc =
                    wpack + (size_t)((kh * TAPS + kw) * KB * NT) * 1024;
                #pragma unroll
                for (int off = 0; off < CHUNK; off += 256 * 16) {
                    int o = off + threadIdx.x * 16;
#if USE_ASYNC_COPY
                    __builtin_amdgcn_global_load_async_to_lds_b128(
                        (gas_v4i*)(gsrc + o), (las_v4i*)(sB + o), 0, 0);
#else
                    *(v4i*)(sB + o) = *(const v4i*)(gsrc + o);
#endif
                }
#if USE_ASYNC_COPY
                __builtin_amdgcn_s_wait_asynccnt(0);
#endif
                __syncthreads();
            }

            int w_in = (w0 + m) * STRIDE + kw - PAD;
            bool ok = h_ok && (w_in >= 0) && (w_in < Win);
            const int* ap =
                (const int*)(rowbase + (size_t)(ok ? w_in : 0) * Cin) + khalf * 8;
            #pragma unroll
            for (int kb = 0; kb < KB; ++kb) {
                v8i a = {0, 0, 0, 0, 0, 0, 0, 0};
                if (ok) {
                    const int* p = ap + kb * 16;
                    #pragma unroll
                    for (int i = 0; i < 8; ++i) a[i] = p[i];
                }
                const v8i* bp = (const v8i*)sB + (size_t)(kb * NT) * 32;
                #pragma unroll
                for (int ct = 0; ct < NT; ++ct) {
                    v8i b = bp[ct * 32 + lane];
                    acc[ct] = __builtin_amdgcn_wmma_i32_16x16x64_iu8(
                                  false, a, true, b, acc[ct], false, false);
                }
            }
        }
    }

    const int nl    = lane & 15;
    const int mbase = khalf * 8;
    if constexpr (LAST) {
        #pragma unroll
        for (int ct = 0; ct < NT; ++ct) {
            int coutg = ct * 16 + nl;
            float sc = scale[coutg], sh = shift[coutg];
            float sum = 0.f;
            #pragma unroll
            for (int r = 0; r < 8; ++r)
                sum += fmaxf((float)acc[ct][r] * sc + sh, 0.f);
            atomicAdd(&pooled[n * Cout + coutg], sum);
        }
    } else {
        int8_t* obase = act_out + ((size_t)(n * Hout + h) * Wout) * Cout;
        #pragma unroll
        for (int ct = 0; ct < NT; ++ct) {
            int coutg = ct * 16 + nl;
            int cperm = permch(coutg);
            float sc = scale[coutg], sh = shift[coutg];
            #pragma unroll
            for (int r = 0; r < 8; ++r) {
                float y = (float)acc[ct][r] * sc + sh;
                obase[(size_t)(w0 + mbase + r) * Cout + cperm] =
                    (y > 0.f) ? (int8_t)1 : (int8_t)0;
            }
        }
    }
}

__global__ void zero_kernel(float* p, int n)
{
    int i = blockIdx.x * blockDim.x + threadIdx.x;
    if (i < n) p[i] = 0.f;
}

__global__ void tail_kernel(const float* __restrict__ pooled,
                            const float* __restrict__ c6w, const float* __restrict__ c6b,
                            const float* __restrict__ fcw, const float* __restrict__ fcb,
                            float* __restrict__ out)
{
    int n = blockIdx.x * blockDim.x + threadIdx.x;
    if (n >= 128) return;
    const float* s = pooled + n * 192;
    float t[12];
    for (int k = 0; k < 12; ++k) {
        float a = 0.f;
        for (int c = 0; c < 192; ++c) a = fmaf(c6w[k * 192 + c], s[c], a);
        t[k] = a * (1.f / 1024.f) + c6b[k];
    }
    float lg[12], mx = -1e30f;
    for (int j = 0; j < 12; ++j) {
        float a = fcb[j];
        for (int k = 0; k < 12; ++k) a = fmaf(fcw[j * 12 + k], t[k], a);
        lg[j] = a; mx = fmaxf(mx, a);
    }
    float sum = 0.f;
    for (int j = 0; j < 12; ++j) { lg[j] = __expf(lg[j] - mx); sum += lg[j]; }
    float inv = 1.f / sum;
    for (int j = 0; j < 12; ++j) out[n * 12 + j] = lg[j] * inv;
}

extern "C" void kernel_launch(void* const* d_in, const int* in_sizes, int n_in,
                              void* d_out, int out_size, void* d_ws, size_t ws_size,
                              hipStream_t stream)
{
    const float* x   = (const float*)d_in[0];
    const float* c1w = (const float*)d_in[1];
    const float* c1b = (const float*)d_in[2];
    const float* w2  = (const float*)d_in[3];
    const float* w3  = (const float*)d_in[4];
    const float* w4  = (const float*)d_in[5];
    const float* w5  = (const float*)d_in[6];
    const float* c6w = (const float*)d_in[7];
    const float* c6b = (const float*)d_in[8];
    const float* fcw = (const float*)d_in[9];
    const float* fcb = (const float*)d_in[10];
    const float* bn[20];
    for (int i = 0; i < 20; ++i) bn[i] = (const float*)d_in[11 + i];

    uint8_t* ws = (uint8_t*)d_ws;
    float*  bn_scale = (float*)(ws + WS_BN_SCALE);
    float*  bn_shift = (float*)(ws + WS_BN_SHIFT);
    float*  pooled   = (float*)(ws + WS_POOLED);
    int8_t* W2 = (int8_t*)(ws + WS_W2);
    int8_t* W3 = (int8_t*)(ws + WS_W3);
    int8_t* W4 = (int8_t*)(ws + WS_W4);
    int8_t* W5 = (int8_t*)(ws + WS_W5);
    int8_t* A2 = (int8_t*)(ws + WS_A2);
    int8_t* A3 = (int8_t*)(ws + WS_A3);
    int8_t* A4 = (int8_t*)(ws + WS_A4);
    int8_t* A5 = (int8_t*)(ws + WS_A5);

    bn_prep_kernel<<<4, 256, 0, stream>>>(
        bn[0], bn[1], bn[2], bn[3],  bn[4], bn[5], bn[6], bn[7],
        bn[8], bn[9], bn[10], bn[11], bn[12], bn[13], bn[14], bn[15],
        bn[16], bn[17], bn[18], bn[19], bn_scale, bn_shift);

    pack_weights_kernel<<< 9, 256, 0, stream>>>(w2, W2,  64, 128, 3);
    pack_weights_kernel<<<18, 256, 0, stream>>>(w3, W3, 128, 128, 3);
    pack_weights_kernel<<<27, 256, 0, stream>>>(w4, W4, 128, 192, 3);
    pack_weights_kernel<<< 5, 256, 0, stream>>>(w5, W5, 192, 192, 1);

    zero_kernel<<<96, 256, 0, stream>>>(pooled, 128 * 192);

    conv1_kernel<<<2048, 256, 0, stream>>>(x, c1w, c1b, bn_scale, bn_shift, A2);

    binconv_wmma<64, 64, 64, 64, 64, 128, 1, 1, 3, false>
        <<<4096, 256, 0, stream>>>(A2, A3, W2, bn_scale + 192, bn_shift + 192, nullptr);
    binconv_wmma<64, 64, 128, 32, 32, 128, 2, 1, 3, false>
        <<<1024, 256, 0, stream>>>(A3, A4, W3, bn_scale + 384, bn_shift + 384, nullptr);
    binconv_wmma<32, 32, 128, 32, 32, 192, 1, 1, 3, false>
        <<<1024, 256, 0, stream>>>(A4, A5, W4, bn_scale + 576, bn_shift + 576, nullptr);
    binconv_wmma<32, 32, 192, 32, 32, 192, 1, 0, 1, true>
        <<<1024, 256, 0, stream>>>(A5, nullptr, W5, bn_scale + 768, bn_shift + 768, pooled);

    tail_kernel<<<1, 128, 0, stream>>>(pooled, c6w, c6b, fcw, fcb, (float*)d_out);
}
